// DispatchGNN_38783554683458
// MI455X (gfx1250) — compile-verified, matrix operand order checked by reading, not run on hardware
//
#include <hip/hip_runtime.h>
#include <hip/hip_bf16.h>
#include <cstdint>
#include <cstddef>

// ---------------------------------------------------------------------------
// CDNA5 (gfx1250) wave32 WMMA types
// ---------------------------------------------------------------------------
typedef __bf16 bf16;
typedef __bf16 v16bf __attribute__((ext_vector_type(16)));
typedef float  v8f   __attribute__((ext_vector_type(8)));
typedef unsigned int u32x4 __attribute__((ext_vector_type(4)));
typedef int i32x4 __attribute__((ext_vector_type(4)));
typedef int i32x8 __attribute__((ext_vector_type(8)));

#define WMMA_BF16(A, B, C) \
  __builtin_amdgcn_wmma_f32_16x16x32_bf16(false, (A), false, (B), (short)0, (C), false, false)

#if defined(__has_builtin)
#if __has_builtin(__builtin_amdgcn_tensor_load_to_lds)
#define HAVE_TDM 1
#endif
#endif

__device__ __forceinline__ bf16 tobf(float f) { return (bf16)f; }

// ---------------------------------------------------------------------------
// WMMA fragment layout helpers (CDNA5 ISA 7.12.2, wave32, 16-bit A 16x32):
// lane L: row M = L&15; lanes 0-15 carry K {0..7,16..23}, lanes 16-31 carry
// K {8..15,24..31}; VGPR i<4 -> K=2i,2i+1 (+8 hi half), i>=4 -> K=16+2(i-4).
// ---------------------------------------------------------------------------
__device__ __forceinline__ int frag_kbase(int i) {
  return (i < 4) ? (2 * i) : (16 + 2 * (i - 4));
}

// A fragment from bf16 row-major LDS tile (per-lane chunks are contiguous ->
// ds_load_b128 pairs).
__device__ __forceinline__ v16bf load_a_frag_lds(const bf16* A, int lda,
                                                 int row0, int k0, int lane) {
  const bf16* p = A + (row0 + (lane & 15)) * lda + k0 + (lane >> 4) * 8;
  v16bf a;
#pragma unroll
  for (int i = 0; i < 8; ++i) {
    const int kb = frag_kbase(i);
    a[2 * i]     = p[kb];
    a[2 * i + 1] = p[kb + 1];
  }
  return a;
}

// A fragment from f32 row-major LDS tile (one 32-wide k-tile), convert in regs.
__device__ __forceinline__ v16bf load_a_frag_lds_f32(const float* A, int lda,
                                                     int row0, int lane) {
  const float* p = A + (row0 + (lane & 15)) * lda + (lane >> 4) * 8;
  v16bf a;
#pragma unroll
  for (int i = 0; i < 8; ++i) {
    const int kb = frag_kbase(i);
    a[2 * i]     = tobf(p[kb]);
    a[2 * i + 1] = tobf(p[kb + 1]);
  }
  return a;
}

// A fragment straight from global f32 row-major (contiguous per-lane chunks).
__device__ __forceinline__ v16bf load_a_frag_global_f32(const float* A, int lda,
                                                        int row0, int k0, int lane,
                                                        int rowmax) {
  const int row  = row0 + (lane & 15);
  const bool rok = row < rowmax;
  const float* p = A + (size_t)(rok ? row : 0) * lda + k0 + (lane >> 4) * 8;
  v16bf a;
#pragma unroll
  for (int i = 0; i < 8; ++i) {
    const int kb = frag_kbase(i);
    float f0 = rok ? p[kb]     : 0.0f;
    float f1 = rok ? p[kb + 1] : 0.0f;
    a[2 * i]     = tobf(f0);
    a[2 * i + 1] = tobf(f1);
  }
  return a;
}

// B fragments are pre-swizzled into LDS so each lane's 16 bf16 (32B) are
// contiguous: layout [kt][nt][lane][16].  Load = single v16bf (2x ds_load_b128).
__device__ __forceinline__ v16bf load_b_frag_sw(const bf16* Wf, int kt, int nnt,
                                                int nt, int lane) {
  return *(const v16bf*)(Wf + (((kt * nnt) + nt) * 32 + lane) * 16);
}

// Stage a f32 row-major weight matrix from global into fragment-order bf16 LDS.
__device__ __forceinline__ void stage_w_frags(bf16* __restrict__ dst,
                                              const float* __restrict__ W, int ldb,
                                              int nkt, int nnt, int k0base,
                                              int kmax, int nmax, int tid, int nth) {
  const int total = nkt * nnt * 32;
  for (int f = tid; f < total; f += nth) {
    const int kt = f / (nnt * 32);
    const int nt = (f / 32) % nnt;
    const int l  = f & 31;
    const int col = nt * 16 + (l & 15);
    const int kh  = (l >> 4) * 8;
    const bool cok = col < nmax;
    bf16* d = dst + f * 16;
#pragma unroll
    for (int i = 0; i < 8; ++i) {
      const int kb = frag_kbase(i);
      const int k  = k0base + kt * 32 + kh + kb;
      d[2 * i]     = (cok && k     < kmax) ? tobf(W[(size_t)k * ldb + col])       : tobf(0.0f);
      d[2 * i + 1] = (cok && k + 1 < kmax) ? tobf(W[(size_t)(k + 1) * ldb + col]) : tobf(0.0f);
    }
  }
}

// ---------------------------------------------------------------------------
// K1: deg = 1 (self loop), mask = 0, cnt = 0
// ---------------------------------------------------------------------------
__global__ void k_init(float* deg, int* mask, int* cnt, int n) {
  int i = blockIdx.x * blockDim.x + threadIdx.x;
  if (i == 0) *cnt = 0;
  if (i < n) { deg[i] = 1.0f; mask[i] = 0; }
}

__global__ void k_mark(const int* __restrict__ tgt, int* __restrict__ mask, int t) {
  int i = blockIdx.x * blockDim.x + threadIdx.x;
  if (i < t) mask[tgt[i]] = 1;
}

__global__ void k_deg(const int* __restrict__ ei, int e, float* __restrict__ deg) {
  int i = blockIdx.x * blockDim.x + threadIdx.x;
  if (i < e) unsafeAtomicAdd(&deg[ei[e + i]], 1.0f);  // dst row of edge_index
}

__global__ void k_rsqrt(float* d, int n) {
  int i = blockIdx.x * blockDim.x + threadIdx.x;
  if (i < n) d[i] = rsqrtf(d[i]);
}

// ---------------------------------------------------------------------------
// K5: xw = x @ W_gcn  [N,128]@[128,128].  8 waves, 128 rows/block.
// W pre-swizzled to fragment order (32KB); x tile staged per 32-wide k-tile
// via the Tensor Data Mover (padded to a 36-float LDS row stride), fallback
// to a cooperative copy when the TDM builtin is unavailable.
// ---------------------------------------------------------------------------
__global__ __launch_bounds__(256) void k_xw_gemm(const float* __restrict__ x,
                                                 const float* __restrict__ W,
                                                 float* __restrict__ xw, int n) {
  __shared__ bf16  Wf[4 * 8 * 32 * 16] __attribute__((aligned(32)));  // 32 KB
  __shared__ float Xlf[128 * 36] __attribute__((aligned(16)));        // 18 KB
  const int tid  = threadIdx.x;
  const int lane = tid & 31;
  const int wv   = tid >> 5;
  const int row0 = blockIdx.x * 128;

  stage_w_frags(Wf, W, 128, 4, 8, 0, 128, 128, tid, 256);

  int rows = n - row0;
  if (rows > 128) rows = 128;

  v8f acc[8] = {};
  for (int kt = 0; kt < 4; ++kt) {
    __syncthreads();  // Xlf reuse + (first iter) Wf staging visibility
#if defined(HAVE_TDM)
    if (wv == 0) {
      const unsigned long long ga =
          (unsigned long long)(uintptr_t)(x + (size_t)row0 * 128 + kt * 32);
      u32x4 g0;
      g0[0] = 1u;                                   // count = 1
      g0[1] = (unsigned)(uintptr_t)(void*)Xlf;      // LDS offset = flat addr[31:0]
      g0[2] = (unsigned)ga;                         // global_addr lo
      g0[3] = (unsigned)((ga >> 32) & 0x01FFFFFFu) | 0x80000000u;  // addr hi | type=2
      i32x8 g1 = {};
      g1[0] = (2 << 16) | (1 << 20) | (4 << 22) | (3 << 25);  // 4B elems, pad 4dw/128B
      g1[1] = 32 << 16;        // tensor_dim0 = 32
      g1[2] = rows << 16;      // tensor_dim1 = remaining rows (OOB rows -> 0)
      g1[3] = 32 << 16;        // tile_dim0 = 32
      g1[4] = 128;             // tile_dim1 = 128
      g1[5] = 128;             // tensor_dim0_stride = 128
      i32x4 z4 = {};
#if __clang_major__ >= 23
      i32x8 z8 = {};
      __builtin_amdgcn_tensor_load_to_lds(g0, g1, z4, z4, z8, 0);
#else
      __builtin_amdgcn_tensor_load_to_lds(g0, g1, z4, z4, 0);
#endif
      __builtin_amdgcn_s_wait_tensorcnt(0);
    }
#else
    for (int idx = tid; idx < 128 * 32; idx += 256) {
      int m = idx >> 5, k = idx & 31;
      Xlf[m * 36 + k] = (m < rows) ? x[(size_t)(row0 + m) * 128 + kt * 32 + k] : 0.0f;
    }
#endif
    __syncthreads();
    v16bf a = load_a_frag_lds_f32(Xlf, 36, wv * 16, lane);
#pragma unroll
    for (int nt = 0; nt < 8; ++nt)
      acc[nt] = WMMA_BF16(a, load_b_frag_sw(Wf, kt, 8, nt, lane), acc[nt]);
  }

  const int cn = lane & 15;
  const int mh = (lane >> 4) * 8;
#pragma unroll
  for (int nt = 0; nt < 8; ++nt)
#pragma unroll
    for (int i = 0; i < 8; ++i) {
      int gr = row0 + wv * 16 + i + mh;
      if (gr < n) xw[(size_t)gr * 128 + nt * 16 + cn] = acc[nt][i];
    }
}

// ---------------------------------------------------------------------------
// K6: agg[i] = xw[i] * dinv[i]^2 for target nodes (self-loop term)
// ---------------------------------------------------------------------------
__global__ void k_agg_init(const float* __restrict__ xw, const float* __restrict__ dinv,
                           const int* __restrict__ mask, float* __restrict__ agg, int n) {
  int idx = blockIdx.x * blockDim.x + threadIdx.x;
  int node = idx >> 5, q = idx & 31;
  if (node < n && mask[node]) {
    float s = dinv[node]; s = s * s;
    const float4 v = *(const float4*)(xw + (size_t)node * 128 + q * 4);
    float4 r = {v.x * s, v.y * s, v.z * s, v.w * s};
    *(float4*)(agg + (size_t)node * 128 + q * 4) = r;
  }
}

// ---------------------------------------------------------------------------
// K7a: compact edges whose dst is a target (warp-aggregated, wave32)
// ---------------------------------------------------------------------------
__global__ void k_compact(const int* __restrict__ ei, int e, const int* __restrict__ mask,
                          int* __restrict__ cnt, int2* __restrict__ pairs) {
  int i = blockIdx.x * blockDim.x + threadIdx.x;
  bool pred = false; int s = 0, d = 0;
  if (i < e) { s = ei[i]; d = ei[e + i]; pred = (mask[d] != 0); }
  unsigned long long bal = __ballot(pred);
  if (bal) {
    int lane = threadIdx.x & 31;
    int leader = __ffsll((long long)bal) - 1;
    int base = 0;
    if (lane == leader) base = atomicAdd(cnt, (int)__popcll(bal));
    base = __shfl(base, leader, 32);
    if (pred) {
      int off = (int)__popcll(bal & ((1ull << lane) - 1ull));
      pairs[base + off] = make_int2(s, d);
    }
  }
}

// ---------------------------------------------------------------------------
// K7b: scatter messages for compacted edges, one wave per edge, persistent.
// ---------------------------------------------------------------------------
__global__ __launch_bounds__(256) void k_scatter(const int2* __restrict__ pairs,
                                                 const int* __restrict__ cnt,
                                                 const float* __restrict__ dinv,
                                                 const float* __restrict__ xw,
                                                 float* __restrict__ agg) {
  const int lane = threadIdx.x & 31;
  const int wave = (blockIdx.x * blockDim.x + threadIdx.x) >> 5;
  const int nw   = (gridDim.x * blockDim.x) >> 5;
  const int n    = *cnt;
  for (int e = wave; e < n; e += nw) {
    __builtin_prefetch(&pairs[e + nw], 0, 1);  // global_prefetch_b8
    int2 p = pairs[e];
    float norm = dinv[p.x] * dinv[p.y];
    const float4 v = *(const float4*)(xw + (size_t)p.x * 128 + lane * 4);
    float* dst = agg + (size_t)p.y * 128 + lane * 4;
    unsafeAtomicAdd(dst + 0, v.x * norm);
    unsafeAtomicAdd(dst + 1, v.y * norm);
    unsafeAtomicAdd(dst + 2, v.z * norm);
    unsafeAtomicAdd(dst + 3, v.w * norm);
  }
}

// ---------------------------------------------------------------------------
// K9: fused truck head.  Block = 64 threads = 2 waves, 32 trucks/block.
// All weight matrices staged to fragment-order LDS (32 KB buffer, reused).
//  A1: truck_emb = relu(truck_x @ W_truck + b_truck)   -> Cin[:,0:128]
//  A2: gather relu(agg[target] + b_gcn)                -> Cin[:,128:256]
//  B : comb = relu(Cin @ W_comb + b_comb)  (K=256 in two staged halves)
//  C : out  = comb @ W_out + b_out         (K=128, N=100 padded to 112)
// ---------------------------------------------------------------------------
__global__ __launch_bounds__(64) void k_truck_head(
    const float* __restrict__ truck_x, const int* __restrict__ tgt,
    const float* __restrict__ W_truck, const float* __restrict__ b_truck,
    const float* __restrict__ W_comb,  const float* __restrict__ b_comb,
    const float* __restrict__ W_out,   const float* __restrict__ b_out,
    const float* __restrict__ agg,     const float* __restrict__ b_gcn,
    float* __restrict__ out, int T, int n_cls) {
  __shared__ bf16 Wbuf[4 * 8 * 32 * 16] __attribute__((aligned(32)));  // 32 KB
  __shared__ bf16 Cin[32 * 264] __attribute__((aligned(16)));          // 16.9 KB
  __shared__ bf16 Mid[32 * 136] __attribute__((aligned(16)));          // 8.7 KB
  const int tid  = threadIdx.x;
  const int lane = tid & 31;
  const int wv   = tid >> 5;
  const int t0   = blockIdx.x * 32;
  const int tw0  = t0 + wv * 16;
  const int cn   = lane & 15;
  const int mh   = (lane >> 4) * 8;

  // --- Phase A1: truck embedding GEMM -> Cin[:, 0:128]
  stage_w_frags(Wbuf, W_truck, 128, 2, 8, 0, 64, 128, tid, 64);
  __syncthreads();
  {
    v8f acc[8] = {};
#pragma unroll
    for (int kt = 0; kt < 2; ++kt) {
      v16bf a = load_a_frag_global_f32(truck_x, 64, tw0, kt * 32, lane, T);
#pragma unroll
      for (int nt = 0; nt < 8; ++nt)
        acc[nt] = WMMA_BF16(a, load_b_frag_sw(Wbuf, kt, 8, nt, lane), acc[nt]);
    }
#pragma unroll
    for (int nt = 0; nt < 8; ++nt)
#pragma unroll
      for (int i = 0; i < 8; ++i) {
        int ml = wv * 16 + i + mh;
        int c  = nt * 16 + cn;
        Cin[ml * 264 + c] = tobf(fmaxf(acc[nt][i] + b_truck[c], 0.0f));
      }
  }

  // --- Phase A2: gather node embeddings for targets -> Cin[:, 128:256]
  for (int idx = tid; idx < 32 * 128; idx += 64) {
    int tl = idx >> 7, h = idx & 127;
    int tg = t0 + tl;
    float v = 0.0f;
    if (tg < T) {
      int node = tgt[tg];
      v = fmaxf(agg[(size_t)node * 128 + h] + b_gcn[h], 0.0f);
    }
    Cin[tl * 264 + 128 + h] = tobf(v);
  }

  // --- Phase B: comb GEMM, K = 256 staged in two 128-deep halves
  {
    v8f acc[8] = {};
#pragma unroll
    for (int half = 0; half < 2; ++half) {
      __syncthreads();  // Wbuf reuse + Cin ready (first half)
      stage_w_frags(Wbuf, W_comb, 128, 4, 8, half * 128, 256, 128, tid, 64);
      __syncthreads();
#pragma unroll
      for (int kt = 0; kt < 4; ++kt) {
        v16bf a = load_a_frag_lds(Cin, 264, wv * 16, half * 128 + kt * 32, lane);
#pragma unroll
        for (int nt = 0; nt < 8; ++nt)
          acc[nt] = WMMA_BF16(a, load_b_frag_sw(Wbuf, kt, 8, nt, lane), acc[nt]);
      }
    }
#pragma unroll
    for (int nt = 0; nt < 8; ++nt)
#pragma unroll
      for (int i = 0; i < 8; ++i) {
        int ml = wv * 16 + i + mh;
        int c  = nt * 16 + cn;
        Mid[ml * 136 + c] = tobf(fmaxf(acc[nt][i] + b_comb[c], 0.0f));
      }
  }

  // --- Phase C: output GEMM, K = 128, N = n_cls padded to 112
  __syncthreads();
  stage_w_frags(Wbuf, W_out, n_cls, 4, 7, 0, 128, n_cls, tid, 64);
  __syncthreads();
  {
    v8f acc[7] = {};
#pragma unroll
    for (int kt = 0; kt < 4; ++kt) {
      v16bf a = load_a_frag_lds(Mid, 136, wv * 16, kt * 32, lane);
#pragma unroll
      for (int nt = 0; nt < 7; ++nt)
        acc[nt] = WMMA_BF16(a, load_b_frag_sw(Wbuf, kt, 7, nt, lane), acc[nt]);
    }
#pragma unroll
    for (int nt = 0; nt < 7; ++nt)
#pragma unroll
      for (int i = 0; i < 8; ++i) {
        int c  = nt * 16 + cn;
        int tr = tw0 + i + mh;
        if (c < n_cls && tr < T)
          out[(size_t)tr * n_cls + c] = acc[nt][i] + b_out[c];
      }
  }
}

// ---------------------------------------------------------------------------
// Host launcher
// ---------------------------------------------------------------------------
extern "C" void kernel_launch(void* const* d_in, const int* in_sizes, int n_in,
                              void* d_out, int out_size, void* d_ws, size_t ws_size,
                              hipStream_t stream) {
  const float* x        = (const float*)d_in[0];
  const int*   ei       = (const int*)d_in[1];
  const float* truck_x  = (const float*)d_in[2];
  const int*   tgt      = (const int*)d_in[3];
  const float* W_gcn    = (const float*)d_in[4];
  const float* b_gcn    = (const float*)d_in[5];
  const float* W_truck  = (const float*)d_in[6];
  const float* b_truck  = (const float*)d_in[7];
  const float* W_comb   = (const float*)d_in[8];
  const float* b_comb   = (const float*)d_in[9];
  const float* W_out    = (const float*)d_in[10];
  const float* b_out    = (const float*)d_in[11];

  const int N = in_sizes[0] / 128;  // 100000 nodes
  const int E = in_sizes[1] / 2;    // 1.6M edges
  const int T = in_sizes[2] / 64;   // 4096 trucks
  const int n_cls = out_size / T;   // 100

  auto au = [](size_t v) { return (v + 255) & ~(size_t)255; };
  char* p = (char*)d_ws;
  float* xw   = (float*)p;  p += au((size_t)N * 128 * sizeof(float));
  float* agg  = (float*)p;  p += au((size_t)N * 128 * sizeof(float));
  float* dinv = (float*)p;  p += au((size_t)N * sizeof(float));
  int*   mask = (int*)p;    p += au((size_t)N * sizeof(int));
  int*   cnt  = (int*)p;    p += 256;
  int2*  pairs = (int2*)p;  // worst case E entries

  k_init<<<(N + 255) / 256, 256, 0, stream>>>(dinv, mask, cnt, N);
  k_mark<<<(T + 255) / 256, 256, 0, stream>>>(tgt, mask, T);
  k_deg<<<(E + 255) / 256, 256, 0, stream>>>(ei, E, dinv);
  k_rsqrt<<<(N + 255) / 256, 256, 0, stream>>>(dinv, N);
  k_xw_gemm<<<(N + 127) / 128, 256, 0, stream>>>(x, W_gcn, xw, N);
  k_agg_init<<<((size_t)N * 32 + 255) / 256, 256, 0, stream>>>(xw, dinv, mask, agg, N);
  k_compact<<<(E + 255) / 256, 256, 0, stream>>>(ei, E, mask, cnt, pairs);
  k_scatter<<<1024, 256, 0, stream>>>(pairs, cnt, dinv, xw, agg);
  k_truck_head<<<(T + 31) / 32, 64, 0, stream>>>(
      truck_x, tgt, W_truck, b_truck, W_comb, b_comb, W_out, b_out,
      agg, b_gcn, (float*)d_out, T, n_cls);
}
